// VectorQuantize_17471926960396
// MI455X (gfx1250) — compile-verified, matrix operand order checked by reading, not run on hardware
//
#include <hip/hip_runtime.h>
#include <hip/hip_bf16.h>

// ---------------------------------------------------------------------------
// VQ-VAE vector quantization for MI455X (gfx1250, wave32, WMMA).
// Distances via  s_k = ||c_k||^2 - 2 z.c_k  (||z||^2 constant -> dropped).
// z.c computed by split-bf16 GEMM:  zh*ch + zh*cl + zl*ch  (3x
// v_wmma_f32_16x16x32_bf16), ~fp32 accuracy at 3/8 of f32-WMMA cost.
// Each wave owns 32 rows (2 A-tiles) x 4 codebook col-tiles -> 1.0 load/WMMA.
// K is split over blockIdx.y (8 chunks) for occupancy; partial argmins are
// combined deterministically in the finalize kernel.
// ---------------------------------------------------------------------------

typedef __bf16 bf16_t;
typedef __attribute__((ext_vector_type(8)))  bf16_t v8bf;
typedef __attribute__((ext_vector_type(16))) bf16_t v16bf;
typedef __attribute__((ext_vector_type(8)))  float  v8f;

constexpr int DIM     = 512;    // embedding dim
constexpr int KCB     = 8192;   // codebook entries
constexpr int NROW    = 16384;  // 32*512 encoder vectors
constexpr int KCHUNKS = 8;      // K split across blockIdx.y
constexpr int KCSIZE  = KCB / KCHUNKS;       // 1024 columns per chunk
constexpr int ROWS_PER_BLOCK = 256;          // 8 waves * 32 rows

// ---------------------------------------------------------------------------
// Workspace layout (bytes)
// ---------------------------------------------------------------------------
constexpr size_t WS_XHI   = 0;         // NROW*DIM bf16 = 16 MB
constexpr size_t WS_XLO   = 16777216;  // 16 MB
constexpr size_t WS_CHI   = 33554432;  // KCB*DIM bf16 = 8 MB
constexpr size_t WS_CLO   = 41943040;  // 8 MB
constexpr size_t WS_CNORM = 50331648;  // KCB f32 = 32 KB
constexpr size_t WS_PBS   = 50364416;  // KCHUNKS*NROW f32 = 512 KB
constexpr size_t WS_PBI   = 50888704;  // KCHUNKS*NROW i32 = 512 KB
constexpr size_t WS_PART  = 51412992;  // NROW f32 = 64 KB

// ---------------------------------------------------------------------------
// 1) split x (f32) into bf16 hi + bf16 residual
// ---------------------------------------------------------------------------
__global__ void __launch_bounds__(256)
split_x_kernel(const float* __restrict__ x, bf16_t* __restrict__ hi,
               bf16_t* __restrict__ lo) {
  size_t i = (size_t)blockIdx.x * 256 + threadIdx.x;  // grid sized exactly
  float v = x[i];
  bf16_t h = (bf16_t)v;
  hi[i] = h;
  lo[i] = (bf16_t)(v - (float)h);
}

// ---------------------------------------------------------------------------
// 2) split codebook + compute ||c_k||^2 (one block per codebook row)
// ---------------------------------------------------------------------------
__global__ void __launch_bounds__(256)
cb_prep_kernel(const float* __restrict__ cb, bf16_t* __restrict__ chi,
               bf16_t* __restrict__ clo, float* __restrict__ cnorm) {
  __shared__ float red[256];
  const int k = blockIdx.x;
  const float* row = cb + (size_t)k * DIM;
  float s = 0.0f;
  for (int d = threadIdx.x; d < DIM; d += 256) {
    float v = row[d];
    bf16_t h = (bf16_t)v;
    chi[(size_t)k * DIM + d] = h;
    clo[(size_t)k * DIM + d] = (bf16_t)(v - (float)h);
    s = fmaf(v, v, s);
  }
  red[threadIdx.x] = s;
  __syncthreads();
  for (int off = 128; off > 0; off >>= 1) {
    if (threadIdx.x < off) red[threadIdx.x] += red[threadIdx.x + off];
    __syncthreads();
  }
  if (threadIdx.x == 0) cnorm[k] = red[0];
}

// ---------------------------------------------------------------------------
// Fragment load: 16x32 bf16 A/B fragment per the CDNA5 16-bit layout.
// Lane (l&15) = row/col; lanes 0-15 take K[0..7]+K[16..23], lanes 16-31 take
// K[8..15]+K[24..31]  ->  two disjoint 16-byte pieces from a row-major row.
// ---------------------------------------------------------------------------
__device__ __forceinline__ v16bf load_frag(const bf16_t* __restrict__ rowp,
                                           int d0, int upper) {
  v8bf p0 = *(const v8bf*)(rowp + d0 + upper * 8);
  v8bf p1 = *(const v8bf*)(rowp + d0 + 16 + upper * 8);
  return __builtin_shufflevector(p0, p1, 0, 1, 2, 3, 4, 5, 6, 7, 8, 9, 10, 11,
                                 12, 13, 14, 15);
}

// ---------------------------------------------------------------------------
// 3) WMMA partial argmin. Grid (NROW/256, KCHUNKS) x 256 threads (8 waves).
//    Each wave owns 32 rows (2 A-tiles) and sweeps its 1024-column K chunk
//    in groups of 64 columns (4 tiles of 16).
// ---------------------------------------------------------------------------
__global__ void __launch_bounds__(256)
vq_argmin_kernel(const bf16_t* __restrict__ Xhi, const bf16_t* __restrict__ Xlo,
                 const bf16_t* __restrict__ Chi, const bf16_t* __restrict__ Clo,
                 const float* __restrict__ cnorm, float* __restrict__ pBS,
                 int* __restrict__ pBI) {
  const int lane  = threadIdx.x & 31;
  const int wave  = threadIdx.x >> 5;
  const int r16   = lane & 15;
  const int upper = lane >> 4;
  const int R     = blockIdx.x * ROWS_PER_BLOCK + wave * 32;
  const int kc    = blockIdx.y;
  const int k0    = kc * KCSIZE;

  const bf16_t* aHi0 = Xhi + (size_t)(R + r16) * DIM;
  const bf16_t* aLo0 = Xlo + (size_t)(R + r16) * DIM;
  const bf16_t* aHi1 = Xhi + (size_t)(R + 16 + r16) * DIM;
  const bf16_t* aLo1 = Xlo + (size_t)(R + 16 + r16) * DIM;

  float bestS[2][8];
  int   bestI[2][8];
#pragma unroll
  for (int rt = 0; rt < 2; ++rt)
#pragma unroll
    for (int j = 0; j < 8; ++j) {
      bestS[rt][j] = __builtin_inff();
      bestI[rt][j] = 0x7fffffff;
    }

  for (int kb = k0; kb < k0 + KCSIZE; kb += 64) {
    const bf16_t* bHi[4];
    const bf16_t* bLo[4];
#pragma unroll
    for (int t = 0; t < 4; ++t) {
      int c  = kb + t * 16 + r16;
      bHi[t] = Chi + (size_t)c * DIM;
      bLo[t] = Clo + (size_t)c * DIM;
    }

    v8f acc[2][4];
#pragma unroll
    for (int rt = 0; rt < 2; ++rt)
#pragma unroll
      for (int t = 0; t < 4; ++t) acc[rt][t] = (v8f){0, 0, 0, 0, 0, 0, 0, 0};

#pragma unroll 2
    for (int dc = 0; dc < DIM; dc += 32) {
      v16bf a0h = load_frag(aHi0, dc, upper);
      v16bf a0l = load_frag(aLo0, dc, upper);
      v16bf a1h = load_frag(aHi1, dc, upper);
      v16bf a1l = load_frag(aLo1, dc, upper);
#pragma unroll
      for (int t = 0; t < 4; ++t) {
        v16bf bh = load_frag(bHi[t], dc, upper);
        v16bf bl = load_frag(bLo[t], dc, upper);
        acc[0][t] = __builtin_amdgcn_wmma_f32_16x16x32_bf16(
            false, a0h, false, bh, (short)0, acc[0][t], false, false);
        acc[0][t] = __builtin_amdgcn_wmma_f32_16x16x32_bf16(
            false, a0h, false, bl, (short)0, acc[0][t], false, false);
        acc[0][t] = __builtin_amdgcn_wmma_f32_16x16x32_bf16(
            false, a0l, false, bh, (short)0, acc[0][t], false, false);
        acc[1][t] = __builtin_amdgcn_wmma_f32_16x16x32_bf16(
            false, a1h, false, bh, (short)0, acc[1][t], false, false);
        acc[1][t] = __builtin_amdgcn_wmma_f32_16x16x32_bf16(
            false, a1h, false, bl, (short)0, acc[1][t], false, false);
        acc[1][t] = __builtin_amdgcn_wmma_f32_16x16x32_bf16(
            false, a1l, false, bh, (short)0, acc[1][t], false, false);
      }
    }

    // Prefetch next K-group's codebook rows into cache (global_prefetch_b8).
    if (kb + 64 < k0 + KCSIZE) {
      __builtin_prefetch(Chi + (size_t)(kb + 64 + r16) * DIM, 0, 0);
      __builtin_prefetch(Clo + (size_t)(kb + 64 + r16) * DIM, 0, 0);
    }

    // Fold the 32x64 score tile into the per-lane running argmin.
    // C layout: lane l, VGPR j -> (row = j + 8*(l>>4), col = l&15).
#pragma unroll
    for (int t = 0; t < 4; ++t) {
      int col  = kb + t * 16 + r16;
      float cn = cnorm[col];
#pragma unroll
      for (int rt = 0; rt < 2; ++rt)
#pragma unroll
        for (int j = 0; j < 8; ++j) {
          float s = fmaf(-2.0f, acc[rt][t][j], cn);
          if (s < bestS[rt][j] || (s == bestS[rt][j] && col < bestI[rt][j])) {
            bestS[rt][j] = s;
            bestI[rt][j] = col;
          }
        }
    }
  }

  // Cross-lane min over the 16 lanes sharing each row (first-index tiebreak).
#pragma unroll
  for (int m = 1; m < 16; m <<= 1) {
#pragma unroll
    for (int rt = 0; rt < 2; ++rt)
#pragma unroll
      for (int j = 0; j < 8; ++j) {
        float os = __shfl_xor(bestS[rt][j], m, 32);
        int   oi = __shfl_xor(bestI[rt][j], m, 32);
        if (os < bestS[rt][j] ||
            (os == bestS[rt][j] && oi < bestI[rt][j])) {
          bestS[rt][j] = os;
          bestI[rt][j] = oi;
        }
      }
  }
  if (r16 == 0) {
#pragma unroll
    for (int rt = 0; rt < 2; ++rt)
#pragma unroll
      for (int j = 0; j < 8; ++j) {
        int row = R + rt * 16 + upper * 8 + j;
        pBS[(size_t)kc * NROW + row] = bestS[rt][j];
        pBI[(size_t)kc * NROW + row] = bestI[rt][j];
      }
  }
}

// ---------------------------------------------------------------------------
// 4) combine K-chunk partials, gather z_q, write indices (as float),
//    per-row squared-diff partial sums
// ---------------------------------------------------------------------------
__global__ void __launch_bounds__(256)
finalize_kernel(const float* __restrict__ x, const float* __restrict__ cb,
                const float* __restrict__ pBS, const int* __restrict__ pBI,
                float* __restrict__ out_zq, float* __restrict__ out_idx,
                float* __restrict__ partial) {
  __shared__ float red[256];
  const int n = blockIdx.x;

  // Deterministic argmin across the KCHUNKS partials (same in every thread).
  float bs = __builtin_inff();
  int idx  = 0x7fffffff;
#pragma unroll
  for (int c = 0; c < KCHUNKS; ++c) {
    float s = pBS[(size_t)c * NROW + n];
    int   i = pBI[(size_t)c * NROW + n];
    if (s < bs || (s == bs && i < idx)) {
      bs  = s;
      idx = i;
    }
  }

  const float* cv = cb + (size_t)idx * DIM;
  const float* z  = x + (size_t)n * DIM;
  float* o = out_zq + (size_t)n * DIM;
  float s = 0.0f;
  for (int d = threadIdx.x; d < DIM; d += 256) {
    float c = cv[d];
    o[d]    = c;  // straight-through forward value == z_q
    float diff = c - z[d];
    s = fmaf(diff, diff, s);
  }
  red[threadIdx.x] = s;
  __syncthreads();
  for (int off = 128; off > 0; off >>= 1) {
    if (threadIdx.x < off) red[threadIdx.x] += red[threadIdx.x + off];
    __syncthreads();
  }
  if (threadIdx.x == 0) {
    partial[n] = red[0];
    out_idx[n] = (float)idx;
  }
}

// ---------------------------------------------------------------------------
// 5) deterministic final reduction: vq_loss == commit_loss (forward values)
// ---------------------------------------------------------------------------
__global__ void __launch_bounds__(256)
loss_reduce_kernel(const float* __restrict__ partial,
                   float* __restrict__ out_loss) {
  __shared__ float red[256];
  float s = 0.0f;
  for (int n = threadIdx.x; n < NROW; n += 256) s += partial[n];
  red[threadIdx.x] = s;
  __syncthreads();
  for (int off = 128; off > 0; off >>= 1) {
    if (threadIdx.x < off) red[threadIdx.x] += red[threadIdx.x + off];
    __syncthreads();
  }
  if (threadIdx.x == 0) {
    float l = red[0] / (float)((size_t)NROW * DIM);
    out_loss[0] = l;  // vq_loss
    out_loss[1] = l;  // commit_loss (same forward value)
  }
}

// ---------------------------------------------------------------------------
extern "C" void kernel_launch(void* const* d_in, const int* in_sizes, int n_in,
                              void* d_out, int out_size, void* d_ws,
                              size_t ws_size, hipStream_t stream) {
  const float* x  = (const float*)d_in[0];  // (32,512,512) f32
  const float* cb = (const float*)d_in[1];  // (8192,512)  f32

  char* ws = (char*)d_ws;
  bf16_t* Xhi  = (bf16_t*)(ws + WS_XHI);
  bf16_t* Xlo  = (bf16_t*)(ws + WS_XLO);
  bf16_t* Chi  = (bf16_t*)(ws + WS_CHI);
  bf16_t* Clo  = (bf16_t*)(ws + WS_CLO);
  float* cnorm = (float*)(ws + WS_CNORM);
  float* pBS   = (float*)(ws + WS_PBS);
  int*   pBI   = (int*)(ws + WS_PBI);
  float* part  = (float*)(ws + WS_PART);

  float* out      = (float*)d_out;
  float* out_zq   = out;                       // NROW*DIM
  float* out_loss = out + (size_t)NROW * DIM;  // [vq_loss, commit_loss]
  float* out_idx  = out_loss + 2;              // NROW indices as float

  split_x_kernel<<<dim3((NROW * DIM) / 256), dim3(256), 0, stream>>>(x, Xhi,
                                                                     Xlo);
  cb_prep_kernel<<<dim3(KCB), dim3(256), 0, stream>>>(cb, Chi, Clo, cnorm);
  vq_argmin_kernel<<<dim3(NROW / ROWS_PER_BLOCK, KCHUNKS), dim3(256), 0,
                     stream>>>(Xhi, Xlo, Chi, Clo, cnorm, pBS, pBI);
  finalize_kernel<<<dim3(NROW), dim3(256), 0, stream>>>(x, cb, pBS, pBI,
                                                        out_zq, out_idx, part);
  loss_reduce_kernel<<<dim3(1), dim3(256), 0, stream>>>(part, out_loss);
}
